// HoMESecondLayer_86955907875073
// MI455X (gfx1250) — matrix operand order repaired and checked by reading, not compiled, not run on hardware
//
#include <hip/hip_runtime.h>
#include <stdint.h>

typedef __attribute__((ext_vector_type(8)))  float        v8f;
typedef __attribute__((ext_vector_type(4)))  float        v4f;
typedef __attribute__((ext_vector_type(4)))  unsigned int v4u;
typedef __attribute__((ext_vector_type(16))) __bf16       v16bf;

#define NB 65536
#define USE_ASYNC_LDS 1

// d_ws offsets (bf16 elements) for pre-swizzled weight fragments
#define OFF_W1  0u          // exp_W1: 6 * (256*128) = 196608
#define OFF_W2  196608u     // exp_W2: 6 * (128*64)  = 49152
#define OFF_W3  245760u     // exp_W3: 6 * (64*64)   = 24576
#define OFF_TG1 270336u     // tg_W1 : 4 * (256*64)  = 65536
#define OFF_TW1 335872u     // tw_W1 : 4 * (64*64)   = 16384
#define OFF_TW2 352256u     // tw_W2 : 4 * (64*32)   = 8192

#if USE_ASYNC_LDS
#define WAIT_ASYNC(n) asm volatile("s_wait_asynccnt " #n ::: "memory")
#else
#define WAIT_ASYNC(n)
#endif

__device__ __forceinline__ float sigf(float x)   { return 1.0f / (1.0f + __expf(-x)); }
__device__ __forceinline__ float swishf(float x) { return x * sigf(x); }

// reduce across the 16-lane half (wave32): masks 1,2,4,8
__device__ __forceinline__ float hred16(float v) {
  v += __shfl_xor(v, 1, 32);
  v += __shfl_xor(v, 2, 32);
  v += __shfl_xor(v, 4, 32);
  v += __shfl_xor(v, 8, 32);
  return v;
}

union FragU { v16bf v; v4u q[2]; };

__device__ __forceinline__ v8f vzero() {
  v8f z;
#pragma unroll
  for (int i = 0; i < 8; ++i) z[i] = 0.0f;
  return z;
}

// ASYNCcnt-tracked copy of 16B: global -> LDS, no VGPR data round-trip.
__device__ __forceinline__ void lds_async_b128(void* lds, const void* g) {
#if USE_ASYNC_LDS
  unsigned l = (unsigned)(uintptr_t)lds;                 // LDS byte addr = addr[31:0]
  unsigned long long ga = (unsigned long long)(uintptr_t)g;
  asm volatile("global_load_async_to_lds_b128 %0, %1, off"
               :: "v"(l), "v"(ga) : "memory");
#else
  *(v4u*)lds = *(const v4u*)g;
#endif
}

// stage `chunks` x 16B from global to LDS across 256 threads
__device__ __forceinline__ void stage_region(__bf16* dst, const __bf16* src,
                                             int chunks, int tid) {
  for (int i = tid; i < chunks; i += 256)
    lds_async_b128((char*)dst + i * 16, (const char*)src + i * 16);
}

// A-fragment (16x32 bf16, MxK): lane = M (l&15), half-wave selects K octets.
__device__ __forceinline__ v16bf load_afrag(const __bf16* base, int row, int stride,
                                            int kk, int lane) {
  const int hlf = lane >> 4;
  const __bf16* p = base + row * stride + kk * 32 + hlf * 8;
  FragU u;
  u.q[0] = *(const v4u*)(p);
  u.q[1] = *(const v4u*)(p + 16);
  return u.v;
}

// B-fragment pre-swizzled: 512 bf16 per frag, lane reads 16 contiguous values
__device__ __forceinline__ v16bf load_bfrag(const __bf16* fragbase, int lane) {
  const __bf16* p = fragbase + lane * 16;
  FragU u;
  u.q[0] = ((const v4u*)p)[0];
  u.q[1] = ((const v4u*)p)[1];
  return u.v;
}

__device__ __forceinline__ v8f wmma_bf16(v16bf a, v16bf b, v8f c) {
  return __builtin_amdgcn_wmma_f32_16x16x32_bf16(false, a, false, b, (short)0, c,
                                                 false, false);
}

// fp32 (K x N row-major, nmat matrices) -> bf16 WMMA-B fragment layout:
// dst[m][kk][nt][lane*16+i] = W[kk*32 + (lane>>4)*8 + (i&7) + (i>>3)*16][nt*16 + (lane&15)]
__global__ void swizzle_w(const float* __restrict__ src, __bf16* __restrict__ dst,
                          int K, int N, int total) {
  int idx = blockIdx.x * 256 + threadIdx.x;
  if (idx >= total) return;
  int per = K * N;
  int m = idx / per, rem = idx % per;
  int frag = rem >> 9, li = rem & 511;
  int lane = li >> 4, i = li & 15;
  int ntiles = N >> 4;
  int kk = frag / ntiles, nt = frag % ntiles;
  int k = kk * 32 + (lane >> 4) * 8 + (i & 7) + (i >> 3) * 16;
  int n = nt * 16 + (lane & 15);
  dst[idx] = (__bf16)src[m * per + k * N + n];
}

__global__ __launch_bounds__(256) void home_main(
    const float* __restrict__ zs,  const float* __restrict__ zg0,
    const float* __restrict__ zg1,
    const float* __restrict__ eb1, const float* __restrict__ eb2,
    const float* __restrict__ eb3, const float* __restrict__ lns,
    const float* __restrict__ lnb, const float* __restrict__ fgA,
    const float* __restrict__ fgB, const float* __restrict__ tgb1,
    const float* __restrict__ tgW2,const float* __restrict__ tgb2,
    const float* __restrict__ sgW, const float* __restrict__ sgb,
    const float* __restrict__ twb1,const float* __restrict__ twb2,
    const float* __restrict__ twW3,const float* __restrict__ twb3,
    const __bf16* __restrict__ ws, float* __restrict__ out) {
  extern __shared__ __align__(16) char smem_raw[];
  __bf16* cmL  = (__bf16*)smem_raw;     // 128 x 256        (32768)
  __bf16* W1B0 = cmL  + 32768;          // W1 ping / tg_W1  (32768)
  __bf16* W1B1 = W1B0 + 32768;          // W1 pong          (32768)
  __bf16* W2B0 = W1B1 + 32768;          // W2 ping          (8192)
  __bf16* W2B1 = W2B0 + 8192;           // W2 pong          (8192)
  __bf16* W3B0 = W2B1 + 8192;           // W3 ping          (4096)
  __bf16* W3B1 = W3B0 + 4096;           // W3 pong          (4096)
  __bf16* h1L  = W3B1 + 4096;           // 128 x 128        (16384)
  __bf16* h2L  = h1L  + 16384;          // 128 x 64         (8192)
  __bf16* aggL = h2L  + 8192;           // 128 x 64         (8192)
  float*  red  = (float*)(aggL + 8192); // 128 x 16 f32 gate partials

  const int t = blockIdx.y;
  const int g = t >> 1;
  const float* zg = g ? zg1 : zg0;
  const int row0 = blockIdx.x * 128;
  const int tid = threadIdx.x;

  // Kick off async weight staging immediately: tg_W1 first (waited earliest),
  // then expert-0 weights (ei==0 for every task) -> overlap with gate math.
  stage_region(W1B0, ws + OFF_TG1 + t * 16384, 2048, tid);  // 8 chunks/thread
  stage_region(W1B1, ws + OFF_W1, 4096, tid);               // 16 chunks/thread
  stage_region(W2B0, ws + OFF_W2, 1024, tid);               // 4 chunks/thread
  stage_region(W3B0, ws + OFF_W3, 512,  tid);               // 2 chunks/thread

  // ---------------- Phase A: low-rank gate + gated cm (bf16) ----------------
  {
    const int lr = tid >> 1, hf = tid & 1;   // row 0..127, column half
    const int grow = row0 + lr;
    const float* srcrow = hf ? (zg + grow * 128) : (zs + grow * 128);
    const float* fgAt = fgA + t * 2048 + hf * 1024;
    float tmp[8];
#pragma unroll
    for (int j = 0; j < 8; ++j) tmp[j] = 0.0f;
    for (int c = 0; c < 128; ++c) {
      float raw = srcrow[c];
      const v4f* a4 = (const v4f*)(fgAt + c * 8);
      v4f a0 = a4[0], a1 = a4[1];
      tmp[0] += raw * a0.x; tmp[1] += raw * a0.y;
      tmp[2] += raw * a0.z; tmp[3] += raw * a0.w;
      tmp[4] += raw * a1.x; tmp[5] += raw * a1.y;
      tmp[6] += raw * a1.z; tmp[7] += raw * a1.w;
    }
#pragma unroll
    for (int j = 0; j < 8; ++j) red[lr * 16 + hf * 8 + j] = tmp[j];
    __syncthreads();
    float tf[8];
#pragma unroll
    for (int j = 0; j < 8; ++j) tf[j] = red[lr * 16 + j] + red[lr * 16 + 8 + j];
    const float* fgBt = fgB + t * 2048;
    for (int c = 0; c < 128; ++c) {
      int col = hf * 128 + c;
      float raw = srcrow[c];
      float s = 0.0f;
#pragma unroll
      for (int j = 0; j < 8; ++j) s += tf[j] * fgBt[j * 256 + col];
      float gate = 2.0f * sigf(s);
      cmL[lr * 256 + col] = (__bf16)(raw * gate);
    }
  }
  // async loads complete in order: <=22 outstanding => the 8 tg_W1 chunks done
  WAIT_ASYNC(22);
  __syncthreads();

  const int wv = tid >> 5;
  const int lane = tid & 31;
  const int rb = wv * 16;           // this wave's local row base
  const int hlf = lane >> 4;
  const int lcol = lane & 15;
  const int lrow = rb + lcol;       // local row this lane supplies for A-frags

  // ---------------- Phase B: task gate (gh = swish(cm@tgW1+b), softmax) -----
  float gw[8][4];
  {
    v8f gacc[4] = {vzero(), vzero(), vzero(), vzero()};
#pragma unroll
    for (int kk = 0; kk < 8; ++kk) {
      v16bf a = load_afrag(cmL, lrow, 256, kk, lane);
#pragma unroll
      for (int nt = 0; nt < 4; ++nt)
        gacc[nt] = wmma_bf16(a, load_bfrag(W1B0 + (kk * 4 + nt) * 512, lane), gacc[nt]);
    }
    float logit[8][4];
#pragma unroll
    for (int r = 0; r < 8; ++r)
#pragma unroll
      for (int k = 0; k < 4; ++k) logit[r][k] = 0.0f;
#pragma unroll
    for (int nt = 0; nt < 4; ++nt) {
      const int col = nt * 16 + lcol;
      const float b1v = tgb1[t * 64 + col];
      float w2[4];
#pragma unroll
      for (int k = 0; k < 4; ++k) w2[k] = tgW2[t * 256 + col * 4 + k];
#pragma unroll
      for (int r = 0; r < 8; ++r) {
        float h = swishf(gacc[nt][r] + b1v);
#pragma unroll
        for (int k = 0; k < 4; ++k) logit[r][k] += h * w2[k];
      }
    }
#pragma unroll
    for (int r = 0; r < 8; ++r) {
      float l0 = hred16(logit[r][0]) + tgb2[t * 4 + 0];
      float l1 = hred16(logit[r][1]) + tgb2[t * 4 + 1];
      float l2 = hred16(logit[r][2]) + tgb2[t * 4 + 2];
      float l3 = hred16(logit[r][3]) + tgb2[t * 4 + 3];
      float m = fmaxf(fmaxf(l0, l1), fmaxf(l2, l3));
      float e0 = __expf(l0 - m), e1 = __expf(l1 - m);
      float e2 = __expf(l2 - m), e3 = __expf(l3 - m);
      float inv = 1.0f / (e0 + e1 + e2 + e3);
      gw[r][0] = e0 * inv; gw[r][1] = e1 * inv;
      gw[r][2] = e2 * inv; gw[r][3] = e3 * inv;
    }
  }

  // ---------------- Phase C: experts (double-buffered weight staging) ------
  v8f agg[4] = {vzero(), vzero(), vzero(), vzero()};
#pragma unroll
  for (int e = 0; e < 4; ++e) {
    const int ei = (e < 2) ? e : (e + 2 * g);   // eidx = [0,1,2+2g,3+2g]
    __bf16* W1cur = (e & 1) ? W1B0 : W1B1;      // e0 weights were staged in W1B1
    __bf16* W2cur = (e & 1) ? W2B1 : W2B0;
    __bf16* W3cur = (e & 1) ? W3B1 : W3B0;

    WAIT_ASYNC(0);       // this expert's async stages complete
    __syncthreads();     // ... and previous buffers are no longer being read

    if (e < 3) {         // stream next expert's weights under this one's WMMAs
      const int en = ((e + 1) < 2) ? (e + 1) : (e + 1 + 2 * g);
      __bf16* W1nxt = (e & 1) ? W1B1 : W1B0;
      __bf16* W2nxt = (e & 1) ? W2B0 : W2B1;
      __bf16* W3nxt = (e & 1) ? W3B0 : W3B1;
      stage_region(W1nxt, ws + OFF_W1 + en * 32768, 4096, tid);
      stage_region(W2nxt, ws + OFF_W2 + en * 8192, 1024, tid);
      stage_region(W3nxt, ws + OFF_W3 + en * 4096, 512,  tid);
    }

    // layer1: 16x256 @ 256x128
    v8f h1[8];
#pragma unroll
    for (int nt = 0; nt < 8; ++nt) h1[nt] = vzero();
#pragma unroll
    for (int kk = 0; kk < 8; ++kk) {
      v16bf a = load_afrag(cmL, lrow, 256, kk, lane);
#pragma unroll
      for (int nt = 0; nt < 8; ++nt)
        h1[nt] = wmma_bf16(a, load_bfrag(W1cur + (kk * 8 + nt) * 512, lane), h1[nt]);
    }
#pragma unroll
    for (int nt = 0; nt < 8; ++nt) {
      const int col = nt * 16 + lcol;
      const float bv = eb1[ei * 128 + col];
#pragma unroll
      for (int r = 0; r < 8; ++r)
        h1L[(rb + r + 8 * hlf) * 128 + col] = (__bf16)swishf(h1[nt][r] + bv);
    }

    // layer2: 16x128 @ 128x64
    v8f h2[4] = {vzero(), vzero(), vzero(), vzero()};
#pragma unroll
    for (int kk = 0; kk < 4; ++kk) {
      v16bf a = load_afrag(h1L, lrow, 128, kk, lane);
#pragma unroll
      for (int nt = 0; nt < 4; ++nt)
        h2[nt] = wmma_bf16(a, load_bfrag(W2cur + (kk * 4 + nt) * 512, lane), h2[nt]);
    }
#pragma unroll
    for (int nt = 0; nt < 4; ++nt) {
      const int col = nt * 16 + lcol;
      const float bv = eb2[ei * 64 + col];
#pragma unroll
      for (int r = 0; r < 8; ++r)
        h2L[(rb + r + 8 * hlf) * 64 + col] = (__bf16)swishf(h2[nt][r] + bv);
    }

    // layer3: 16x64 @ 64x64
    v8f ev[4] = {vzero(), vzero(), vzero(), vzero()};
#pragma unroll
    for (int kk = 0; kk < 2; ++kk) {
      v16bf a = load_afrag(h2L, lrow, 64, kk, lane);
#pragma unroll
      for (int nt = 0; nt < 4; ++nt)
        ev[nt] = wmma_bf16(a, load_bfrag(W3cur + (kk * 4 + nt) * 512, lane), ev[nt]);
    }

    // bias + LayerNorm + self-gate scalar + weighted accumulate
    float ex[4][8];
    float lnSv[4], lnBv[4], wsg[4];
#pragma unroll
    for (int nt = 0; nt < 4; ++nt) {
      const int col = nt * 16 + lcol;
      const float b3v = eb3[ei * 64 + col];
      lnSv[nt] = lns[ei * 64 + col];
      lnBv[nt] = lnb[ei * 64 + col];
      wsg[nt]  = sgW[t * 256 + col * 4 + e];
#pragma unroll
      for (int r = 0; r < 8; ++r) ex[nt][r] = ev[nt][r] + b3v;
    }
#pragma unroll
    for (int r = 0; r < 8; ++r) {
      float s  = ex[0][r] + ex[1][r] + ex[2][r] + ex[3][r];
      float s2 = ex[0][r]*ex[0][r] + ex[1][r]*ex[1][r]
               + ex[2][r]*ex[2][r] + ex[3][r]*ex[3][r];
      s = hred16(s); s2 = hred16(s2);
      float mu  = s * 0.015625f;
      float var = s2 * 0.015625f - mu * mu;
      float rs  = rsqrtf(var + 1e-5f);
      float swp = 0.0f;
#pragma unroll
      for (int nt = 0; nt < 4; ++nt) {
        float x = (ex[nt][r] - mu) * rs * lnSv[nt] + lnBv[nt];
        ex[nt][r] = x;
        swp += x * wsg[nt];
      }
      swp = hred16(swp) + sgb[t * 4 + e];
      const float sc = swp * gw[r][e];
#pragma unroll
      for (int nt = 0; nt < 4; ++nt) agg[nt][r] += ex[nt][r] * sc;
    }
  }

  // ---------------- Phase D: tower 64->64->32->1 ---------------------------
#pragma unroll
  for (int nt = 0; nt < 4; ++nt) {
    const int col = nt * 16 + lcol;
#pragma unroll
    for (int r = 0; r < 8; ++r)
      aggL[(rb + r + 8 * hlf) * 64 + col] = (__bf16)agg[nt][r];
  }
  const __bf16* tw1 = ws + OFF_TW1 + t * 4096;  // frags read directly from global
  v8f t1[4] = {vzero(), vzero(), vzero(), vzero()};
#pragma unroll
  for (int kk = 0; kk < 2; ++kk) {
    v16bf a = load_afrag(aggL, lrow, 64, kk, lane);
#pragma unroll
    for (int nt = 0; nt < 4; ++nt)
      t1[nt] = wmma_bf16(a, load_bfrag(tw1 + (kk * 4 + nt) * 512, lane), t1[nt]);
  }
#pragma unroll
  for (int nt = 0; nt < 4; ++nt) {
    const int col = nt * 16 + lcol;
    const float bv = twb1[t * 64 + col];
#pragma unroll
    for (int r = 0; r < 8; ++r)
      h2L[(rb + r + 8 * hlf) * 64 + col] = (__bf16)swishf(t1[nt][r] + bv);
  }
  const __bf16* tw2 = ws + OFF_TW2 + t * 2048;
  v8f t2[2] = {vzero(), vzero()};
#pragma unroll
  for (int kk = 0; kk < 2; ++kk) {
    v16bf a = load_afrag(h2L, lrow, 64, kk, lane);
#pragma unroll
    for (int nt = 0; nt < 2; ++nt)
      t2[nt] = wmma_bf16(a, load_bfrag(tw2 + (kk * 2 + nt) * 512, lane), t2[nt]);
  }
  float w3v[2], b2v[2];
#pragma unroll
  for (int nt = 0; nt < 2; ++nt) {
    w3v[nt] = twW3[t * 32 + nt * 16 + lcol];
    b2v[nt] = twb2[t * 32 + nt * 16 + lcol];
  }
  const float b3t = twb3[t];
#pragma unroll
  for (int r = 0; r < 8; ++r) {
    float p = swishf(t2[0][r] + b2v[0]) * w3v[0]
            + swishf(t2[1][r] + b2v[1]) * w3v[1];
    p = sigf(hred16(p) + b3t);
    if (lcol == 0) out[(row0 + rb + r + 8 * hlf) * 4 + t] = p;
  }
}

extern "C" void kernel_launch(void* const* d_in, const int* in_sizes, int n_in,
                              void* d_out, int out_size, void* d_ws, size_t ws_size,
                              hipStream_t stream) {
  const float* zs   = (const float*)d_in[0];
  const float* zg0  = (const float*)d_in[1];
  const float* zg1  = (const float*)d_in[2];
  // d_in[3] = v (unused by reference)
  const float* eW1  = (const float*)d_in[4];
  const float* eb1  = (const float*)d_in[5];
  const float* eW2  = (const float*)d_in[6];
  const float* eb2  = (const float*)d_in[7];
  const float* eW3  = (const float*)d_in[8];
  const float* eb3  = (const float*)d_in[9];
  const float* lns  = (const float*)d_in[10];
  const float* lnb  = (const float*)d_in[11];
  const float* fgA  = (const float*)d_in[12];
  const float* fgB  = (const float*)d_in[13];
  const float* tgW1 = (const float*)d_in[14];
  const float* tgb1 = (const float*)d_in[15];
  const float* tgW2 = (const float*)d_in[16];
  const float* tgb2 = (const float*)d_in[17];
  const float* sgW  = (const float*)d_in[18];
  const float* sgb  = (const float*)d_in[19];
  const float* twW1 = (const float*)d_in[20];
  const float* twb1 = (const float*)d_in[21];
  const float* twW2 = (const float*)d_in[22];
  const float* twb2 = (const float*)d_in[23];
  const float* twW3 = (const float*)d_in[24];
  const float* twb3 = (const float*)d_in[25];
  (void)in_sizes; (void)n_in; (void)out_size; (void)ws_size;

  __bf16* ws = (__bf16*)d_ws;
  float* out = (float*)d_out;

  auto swz = [&](const float* s, unsigned off, int K, int N, int nm) {
    int total = nm * K * N;
    swizzle_w<<<dim3((total + 255) / 256), dim3(256), 0, stream>>>(s, ws + off, K, N, total);
  };
  swz(eW1,  OFF_W1,  256, 128, 6);
  swz(eW2,  OFF_W2,  128, 64,  6);
  swz(eW3,  OFF_W3,  64,  64,  6);
  swz(tgW1, OFF_TG1, 256, 64,  4);
  swz(twW1, OFF_TW1, 64,  64,  4);
  swz(twW2, OFF_TW2, 64,  32,  4);

  // LDS: cm 64K | W1 x2 128K | W2 x2 32K | W3 x2 16K | h1 32K | h2 16K | agg 16K | red 8K
  const size_t lds_bytes = (size_t)(32768 + 2*32768 + 2*8192 + 2*4096
                                    + 16384 + 8192 + 8192) * 2 + 2048 * 4; // 319488
  home_main<<<dim3(NB / 128, 4), dim3(256), lds_bytes, stream>>>(
      zs, zg0, zg1, eb1, eb2, eb3, lns, lnb, fgA, fgB,
      tgb1, tgW2, tgb2, sgW, sgb, twb1, twb2, twW3, twb3, ws, out);
}
